// SparseConv3d_InPlace_88373247082541
// MI455X (gfx1250) — matrix operand order repaired and checked by reading, not compiled, not run on hardware
//
#include <hip/hip_runtime.h>

typedef __attribute__((ext_vector_type(16))) _Float16 v16h;
typedef __attribute__((ext_vector_type(8)))  float    v8f;
typedef __attribute__((ext_vector_type(4)))  float    v4f;

#define C_IN  64
#define C_OUT 64
#define WAVES_PER_WG 8
#define THREADS (WAVES_PER_WG * 32)

// out[i] = bias[i % 64]
__global__ void spconv_init_out(float* __restrict__ out,
                                const float* __restrict__ bias, int n) {
    int i = blockIdx.x * blockDim.x + threadIdx.x;
    if (i < n) out[i] = bias[i & (C_OUT - 1)];
}

__global__ __launch_bounds__(THREADS) void spconv_wmma(
    const float* __restrict__ x,      // [N_IN, 64]
    const float* __restrict__ w,      // [K, 64, 64]
    const int*   __restrict__ rules,  // [K*RK, 3] (k, in, out)
    float* __restrict__ out,          // [N_OUT, 64]
    int K, int RK, int wgsPerK)
{
    const int lane   = threadIdx.x & 31;
    const int wave   = threadIdx.x >> 5;
    const int lane16 = lane & 15;
    const bool hi    = lane >= 16;

    const int k    = blockIdx.x / wgsPerK;
    const int cidx = blockIdx.x % wgsPerK;
    if (k >= K) return;

    // ---- Preload B fragments (weights for offset k) in WMMA 32x16 layout ----
    // B fragment element j: K = chunk*32 + (hi?16:0) + j, N = nt*16 + lane16
    const float* wk = w + (size_t)k * C_IN * C_OUT;
    v16h B[4][2];
#pragma unroll
    for (int nt = 0; nt < 4; ++nt) {
#pragma unroll
        for (int ch = 0; ch < 2; ++ch) {
            const int kbase = ch * 32 + (hi ? 16 : 0);
            const int n     = nt * 16 + lane16;
            v16h b;
#pragma unroll
            for (int j = 0; j < 16; ++j)
                b[j] = (_Float16)wk[(kbase + j) * C_OUT + n];
            B[nt][ch] = b;
        }
    }

    const int blocksPerOffset = (RK + 15) >> 4;
    const int ruleRowBase     = k * RK;
    const int moff            = hi ? 8 : 0;

    for (int blk = cidx * WAVES_PER_WG + wave; blk < blocksPerOffset;
         blk += wgsPerK * WAVES_PER_WG) {
        const int rbase = blk * 16;

        // gather-row index for this lane's M row (M = lane16)
        int grow = 0;
        {
            int rr = rbase + lane16;
            if (rr < RK) grow = rules[(size_t)(ruleRowBase + rr) * 3 + 1];
        }
        // scatter-row indices for this lane's 8 C rows (M = moff + r)
        int  oidx[8];
        bool oval[8];
#pragma unroll
        for (int r = 0; r < 8; ++r) {
            int rr  = rbase + moff + r;
            oval[r] = rr < RK;
            oidx[r] = oval[r] ? rules[(size_t)(ruleRowBase + rr) * 3 + 2] : 0;
        }

        // ---- Gather A fragments: two 16x32 f16 tiles (K-chunks of 32) ----
        // elements 0..15 : K = abase..abase+15, abase = ch*32 + (hi?16:0)
        const float* xrow = x + (size_t)grow * C_IN;
        v16h A[2];
#pragma unroll
        for (int ch = 0; ch < 2; ++ch) {
            const int abase = ch * 32 + (hi ? 16 : 0);
            v4f f0 = *(const v4f*)(xrow + abase);
            v4f f1 = *(const v4f*)(xrow + abase + 4);
            v4f f2 = *(const v4f*)(xrow + abase + 8);
            v4f f3 = *(const v4f*)(xrow + abase + 12);
            v16h a;
#pragma unroll
            for (int j = 0; j < 4; ++j) {
                a[j]      = (_Float16)f0[j];
                a[4 + j]  = (_Float16)f1[j];
                a[8 + j]  = (_Float16)f2[j];
                a[12 + j] = (_Float16)f3[j];
            }
            A[ch] = a;
        }

        // ---- 4 output tiles of 16 columns; 2 WMMAs each (K = 64) ----
#pragma unroll
        for (int nt = 0; nt < 4; ++nt) {
            v8f acc = {};
            acc = __builtin_amdgcn_wmma_f32_16x16x32_f16(
                false, A[0], false, B[nt][0], (short)0, acc, false, false);
            acc = __builtin_amdgcn_wmma_f32_16x16x32_f16(
                false, A[1], false, B[nt][1], (short)0, acc, false, false);

            const int col = nt * 16 + lane16;
#pragma unroll
            for (int r = 0; r < 8; ++r) {
                if (oval[r]) {
                    float* addr = out + (size_t)oidx[r] * C_OUT + col;
                    float  v    = acc[r];
                    asm volatile("global_atomic_add_f32 %0, %1, off"
                                 :: "v"(addr), "v"(v) : "memory");
                }
            }
        }
    }
}

extern "C" void kernel_launch(void* const* d_in, const int* in_sizes, int n_in,
                              void* d_out, int out_size, void* d_ws, size_t ws_size,
                              hipStream_t stream) {
    const float* x     = (const float*)d_in[0];
    const float* w     = (const float*)d_in[1];
    const float* bias  = (const float*)d_in[2];
    const int*   rules = (const int*)d_in[3];
    float*       out   = (float*)d_out;

    const int K      = in_sizes[1] / (C_IN * C_OUT);
    const int nRules = in_sizes[3] / 3;
    const int RK     = (K > 0) ? nRules / K : 0;

    const int blocksPerOffset = (RK + 15) / 16;
    const int targetBlocksPerWave = 8;
    int wgsPerK = (blocksPerOffset + WAVES_PER_WG * targetBlocksPerWave - 1) /
                  (WAVES_PER_WG * targetBlocksPerWave);
    if (wgsPerK < 1) wgsPerK = 1;

    spconv_init_out<<<(out_size + 255) / 256, 256, 0, stream>>>(out, bias, out_size);
    spconv_wmma<<<K * wgsPerK, THREADS, 0, stream>>>(x, w, rules, out, K, RK, wgsPerK);
}
